// CrossAttentionBlock_35802847379568
// MI455X (gfx1250) — compile-verified, matrix operand order checked by reading, not run on hardware
//
#include <hip/hip_runtime.h>

// ---------------------------------------------------------------------------
// CDNA5 (gfx1250) implementation of the point-cloud cross-attention block.
// All heavy GEMMs run on v_wmma_f32_16x16x32_bf16 (fp32 accumulate).
// ---------------------------------------------------------------------------

typedef __attribute__((ext_vector_type(16))) __bf16 v16bf;
typedef __attribute__((ext_vector_type(8)))  __bf16 v8bf;
typedef __attribute__((ext_vector_type(8)))  float  v8f;

constexpr int Bsz  = 8;
constexpr int Npts = 1024;
constexpr int KNN  = 16;
constexpr int D    = 256;
constexpr int BN   = Bsz * Npts;   // 8192 rows
constexpr int LDSP = 264;          // bf16 LDS row pitch (multiple of 8 -> 16B aligned rows)

static __device__ __forceinline__ __bf16 to_bf(float x) { return (__bf16)x; }

// Load a 16-element bf16 fragment: 8 contiguous at p (K = kb..kb+7) and
// 8 contiguous at p+16 (K = kb+16..kb+23), matching the ISA 16-bit A layout.
static __device__ __forceinline__ v16bf load_frag(const __bf16* p) {
  v8bf lo = *(const v8bf*)(p);
  v8bf hi = *(const v8bf*)(p + 16);
  v16bf r;
#pragma unroll
  for (int i = 0; i < 8; ++i) { r[i] = lo[i]; r[i + 8] = hi[i]; }
  return r;
}

// One 16x16 f32 tile = A[16 x 256] (row pitch ap) * Bt[nb.., 256] via 8 WMMAs.
static __device__ __forceinline__ v8f tile_mm(const __bf16* A, int ap,
                                              const __bf16* Bt, int nb, int lane) {
  const int m  = lane & 15;
  const int kb = (lane >> 4) * 8;
  const __bf16* arow = A + (size_t)m * ap + kb;
  const __bf16* brow = Bt + (size_t)(nb + m) * D + kb;
  v8f c = {};
#pragma unroll
  for (int k0 = 0; k0 < D; k0 += 32) {
    if (k0 + 32 < D) __builtin_prefetch(brow + k0 + 32, 0, 1);
    v16bf a = load_frag(arow + k0);
    v16bf b = load_frag(brow + k0);
    c = __builtin_amdgcn_wmma_f32_16x16x32_bf16(false, a, false, b,
                                                (short)0, c, false, false);
  }
  return c;
}

// ---------------------------------------------------------------------------
// Weight transpose + fp32 -> bf16  (W[k][n] -> Wt[n][k])
// ---------------------------------------------------------------------------
__global__ void wtrans_kernel(const float* __restrict__ W, __bf16* __restrict__ Wt) {
  int n = blockIdx.x * 16 + (threadIdx.x & 15);
  int k = blockIdx.y * 16 + (threadIdx.x >> 4);
  Wt[(size_t)n * D + k] = to_bf(W[(size_t)k * D + n]);
}

__global__ void f2bf_kernel(const float* __restrict__ x, __bf16* __restrict__ y, int n) {
  int i = blockIdx.x * 256 + threadIdx.x;
  if (i < n) y[i] = to_bf(x[i]);
}

// ---------------------------------------------------------------------------
// kNN: one 256-thread block per (b,n); 16 iterative argmin extractions.
// ---------------------------------------------------------------------------
__global__ void __launch_bounds__(256)
knn_kernel(const float* __restrict__ xyz, int* __restrict__ knn) {
  __shared__ float px[Npts], py[Npts], pz[Npts], dist[Npts];
  __shared__ float rv[256];
  __shared__ int   ri[256];
  const int t  = threadIdx.x;
  const int bn = blockIdx.x;
  const int b  = bn >> 10, n = bn & 1023;
  const float* base = xyz + (size_t)b * Npts * 3;
  for (int j = t; j < Npts; j += 256) {
    px[j] = base[j * 3 + 0];
    py[j] = base[j * 3 + 1];
    pz[j] = base[j * 3 + 2];
  }
  __syncthreads();
  const float qx = px[n], qy = py[n], qz = pz[n];
  for (int j = t; j < Npts; j += 256) {
    float dx = qx - px[j], dy = qy - py[j], dz = qz - pz[j];
    dist[j] = dx * dx + dy * dy + dz * dz;
  }
  __syncthreads();
  for (int s = 0; s < KNN; ++s) {
    float bv = 3.4e38f; int bi = 0x7fffffff;
    for (int j = t; j < Npts; j += 256) {
      float d = dist[j];
      if (d < bv || (d == bv && j < bi)) { bv = d; bi = j; }
    }
    rv[t] = bv; ri[t] = bi;
    __syncthreads();
    for (int off = 128; off > 0; off >>= 1) {
      if (t < off) {
        if (rv[t + off] < rv[t] || (rv[t + off] == rv[t] && ri[t + off] < ri[t])) {
          rv[t] = rv[t + off]; ri[t] = ri[t + off];
        }
      }
      __syncthreads();
    }
    if (t == 0) { knn[bn * KNN + s] = ri[0]; dist[ri[0]] = 3.4e38f; }
    __syncthreads();
  }
}

// ---------------------------------------------------------------------------
// Generic WMMA GEMM: C[M,256] = A[M,256](bf16) * Bt(bf16, N-major) (+bias)
// optional relu / fp32 residual / fp32 and-or bf16 stores.
// grid = (M/16, 16), block = 32 (one wave per 16x16 tile).
// ---------------------------------------------------------------------------
__global__ void __launch_bounds__(32)
gemm16_kernel(const __bf16* __restrict__ A, const __bf16* __restrict__ Bt,
              const float* __restrict__ bias, const float* __restrict__ residual,
              float* __restrict__ Cf, __bf16* __restrict__ Cb, int relu_flag) {
  const int lane = threadIdx.x;
  const int tm = blockIdx.x, tn = blockIdx.y;
  v8f c = tile_mm(A + (size_t)tm * 16 * D, D, Bt, tn * 16, lane);
  const int col   = tn * 16 + (lane & 15);
  const int rbase = tm * 16 + (lane >> 4) * 8;
  const float bv = bias ? bias[col] : 0.0f;
#pragma unroll
  for (int i = 0; i < 8; ++i) {
    size_t off = (size_t)(rbase + i) * D + col;
    float v = c[i] + bv;
    if (relu_flag) v = fmaxf(v, 0.0f);
    if (residual) v += residual[off];
    if (Cf) Cf[off] = v;
    if (Cb) Cb[off] = to_bf(v);
  }
}

// ---------------------------------------------------------------------------
// Fused per-(b,n) attention kernel: the 16 neighbors form the M=16 block.
// 256 threads = 8 waves; each wave owns two 16-wide N tiles of D=256.
// Stages: A) pos_hidden=relu(rel@d1+b)  B) pos_enc=@d2+b, h=q-k+pos_enc
//         C) t=relu(h@g1+b)  D) logits=(t@g2+b)/16  E) softmax over K + combine
// ---------------------------------------------------------------------------
__global__ void __launch_bounds__(256)
fused_attn_kernel(const float* __restrict__ xyz, const int* __restrict__ knn,
                  const float* __restrict__ qmat, const float* __restrict__ kall,
                  const float* __restrict__ vall,
                  const float* __restrict__ d1w, const float* __restrict__ d1b,
                  const __bf16* __restrict__ d2t, const float* __restrict__ d2b,
                  const __bf16* __restrict__ g1t, const float* __restrict__ g1b,
                  const __bf16* __restrict__ g2t, const float* __restrict__ g2b,
                  float* __restrict__ attn_out, __bf16* __restrict__ act) {
  __shared__ __align__(16) __bf16 Ab0[16][LDSP];
  __shared__ __align__(16) __bf16 Ab1[16][LDSP];
  __shared__ float pos_s[16][D + 4];
  __shared__ float log_s[16][D + 4];
  __shared__ float q_s[D];
  __shared__ float rel_s[16][3];
  __shared__ int   idx_s[16];

  const int t    = threadIdx.x;
  const int bn   = blockIdx.x;
  const int b    = bn >> 10;
  const int lane = t & 31;
  const int wid  = t >> 5;

  if (t < KNN) idx_s[t] = knn[bn * KNN + t];
  q_s[t] = qmat[(size_t)bn * D + t];
  __syncthreads();
  if (t < KNN) {
    const int j = idx_s[t];
    rel_s[t][0] = xyz[(size_t)bn * 3 + 0] - xyz[(size_t)(b * Npts + j) * 3 + 0];
    rel_s[t][1] = xyz[(size_t)bn * 3 + 1] - xyz[(size_t)(b * Npts + j) * 3 + 1];
    rel_s[t][2] = xyz[(size_t)bn * 3 + 2] - xyz[(size_t)(b * Npts + j) * 3 + 2];
  }
  __syncthreads();

  // Stage A: pos_hidden = relu(rel @ d1 + d1_b) -> Ab0 (bf16)
  {
    const int kk = t >> 4;
    const int fb = (t & 15) * 16;
    const float r0 = rel_s[kk][0], r1 = rel_s[kk][1], r2 = rel_s[kk][2];
#pragma unroll
    for (int j = 0; j < 16; ++j) {
      const int f = fb + j;
      float s = d1b[f] + r0 * d1w[f] + r1 * d1w[D + f] + r2 * d1w[2 * D + f];
      Ab0[kk][f] = to_bf(fmaxf(s, 0.0f));
    }
  }
  __syncthreads();

  // Stage B: pos_enc = Ab0 @ d2t + d2_b ; h = q - k_gather + pos_enc -> Ab1
#pragma unroll
  for (int it = 0; it < 2; ++it) {
    const int nb = (wid * 2 + it) * 16;
    v8f c = tile_mm(&Ab0[0][0], LDSP, d2t, nb, lane);
    const int col = nb + (lane & 15);
    const int rb  = (lane >> 4) * 8;
#pragma unroll
    for (int i = 0; i < 8; ++i) {
      const int row = rb + i;
      const float pos = c[i] + d2b[col];
      pos_s[row][col] = pos;
      const float h = q_s[col] - kall[(size_t)(b * Npts + idx_s[row]) * D + col] + pos;
      Ab1[row][col] = to_bf(h);
    }
  }
  __syncthreads();

  // Stage C: t = relu(Ab1 @ g1t + g1_b) -> Ab0
#pragma unroll
  for (int it = 0; it < 2; ++it) {
    const int nb = (wid * 2 + it) * 16;
    v8f c = tile_mm(&Ab1[0][0], LDSP, g1t, nb, lane);
    const int col = nb + (lane & 15);
    const int rb  = (lane >> 4) * 8;
#pragma unroll
    for (int i = 0; i < 8; ++i)
      Ab0[rb + i][col] = to_bf(fmaxf(c[i] + g1b[col], 0.0f));
  }
  __syncthreads();

  // Stage D: logits = (Ab0 @ g2t + g2_b) / sqrt(D) -> log_s
#pragma unroll
  for (int it = 0; it < 2; ++it) {
    const int nb = (wid * 2 + it) * 16;
    v8f c = tile_mm(&Ab0[0][0], LDSP, g2t, nb, lane);
    const int col = nb + (lane & 15);
    const int rb  = (lane >> 4) * 8;
#pragma unroll
    for (int i = 0; i < 8; ++i)
      log_s[rb + i][col] = (c[i] + g2b[col]) * 0.0625f;
  }
  __syncthreads();

  // Stage E: per feature column f, softmax over the 16 neighbors + combine
  {
    const int f = t;
    float mx = -3.4e38f;
#pragma unroll
    for (int kk = 0; kk < KNN; ++kk) mx = fmaxf(mx, log_s[kk][f]);
    float e[KNN]; float s = 0.0f;
#pragma unroll
    for (int kk = 0; kk < KNN; ++kk) { e[kk] = __expf(log_s[kk][f] - mx); s += e[kk]; }
    const float inv = 1.0f / s;
    float acc = 0.0f;
#pragma unroll
    for (int kk = 0; kk < KNN; ++kk) {
      const float a = e[kk] * inv;
      attn_out[((size_t)bn * KNN + kk) * D + f] = a;
      acc += a * (vall[(size_t)(b * Npts + idx_s[kk]) * D + f] + pos_s[kk][f]);
    }
    act[(size_t)bn * D + f] = to_bf(acc);
  }
}

// ---------------------------------------------------------------------------
// Host-side orchestration
// ---------------------------------------------------------------------------
extern "C" void kernel_launch(void* const* d_in, const int* in_sizes, int n_in,
                              void* d_out, int out_size, void* d_ws, size_t ws_size,
                              hipStream_t stream) {
  (void)in_sizes; (void)n_in; (void)out_size; (void)ws_size;
  const float* xyz   = (const float*)d_in[0];
  const float* sfeat = (const float*)d_in[1];
  const float* tfeat = (const float*)d_in[2];
  const float* fc1_w = (const float*)d_in[3];
  const float* fc1_b = (const float*)d_in[4];
  const float* fc3_w = (const float*)d_in[5];
  const float* fc3_b = (const float*)d_in[6];
  const float* d1_w  = (const float*)d_in[7];
  const float* d1_b  = (const float*)d_in[8];
  const float* d2_w  = (const float*)d_in[9];
  const float* d2_b  = (const float*)d_in[10];
  const float* g1_w  = (const float*)d_in[11];
  const float* g1_b  = (const float*)d_in[12];
  const float* g2_w  = (const float*)d_in[13];
  const float* g2_b  = (const float*)d_in[14];
  const float* wq_w  = (const float*)d_in[15];
  const float* wk_w  = (const float*)d_in[16];
  const float* wv_w  = (const float*)d_in[17];

  char* wsp = (char*)d_ws;
  auto alloc = [&](size_t bytes) -> void* {
    void* p = (void*)wsp;
    wsp += (bytes + 255) & ~(size_t)255;
    return p;
  };
  int*    knn    = (int*)alloc((size_t)BN * KNN * sizeof(int));
  __bf16* srcb   = (__bf16*)alloc((size_t)BN * D * 2);   // search_feat bf16
  __bf16* tmpb   = (__bf16*)alloc((size_t)BN * D * 2);   // template_feat bf16
  __bf16* sfb    = (__bf16*)alloc((size_t)BN * D * 2);   // fc1(search) bf16
  __bf16* tfb    = (__bf16*)alloc((size_t)BN * D * 2);   // fc1(template) bf16
  float*  qmat   = (float*)alloc((size_t)BN * D * 4);
  float*  kall   = (float*)alloc((size_t)BN * D * 4);
  float*  vall   = (float*)alloc((size_t)BN * D * 4);
  __bf16* act    = (__bf16*)alloc((size_t)BN * D * 2);   // attention output (pre-fc3)
  __bf16* fc1t   = (__bf16*)alloc((size_t)D * D * 2);
  __bf16* wqt    = (__bf16*)alloc((size_t)D * D * 2);
  __bf16* wkt    = (__bf16*)alloc((size_t)D * D * 2);
  __bf16* wvt    = (__bf16*)alloc((size_t)D * D * 2);
  __bf16* d2t    = (__bf16*)alloc((size_t)D * D * 2);
  __bf16* g1t    = (__bf16*)alloc((size_t)D * D * 2);
  __bf16* g2t    = (__bf16*)alloc((size_t)D * D * 2);
  __bf16* fc3t   = (__bf16*)alloc((size_t)D * D * 2);

  float* res_out  = (float*)d_out;
  float* attn_out = res_out + (size_t)BN * D;

  const dim3 tg(16, 16);
  wtrans_kernel<<<tg, 256, 0, stream>>>(fc1_w, fc1t);
  wtrans_kernel<<<tg, 256, 0, stream>>>(wq_w,  wqt);
  wtrans_kernel<<<tg, 256, 0, stream>>>(wk_w,  wkt);
  wtrans_kernel<<<tg, 256, 0, stream>>>(wv_w,  wvt);
  wtrans_kernel<<<tg, 256, 0, stream>>>(d2_w,  d2t);
  wtrans_kernel<<<tg, 256, 0, stream>>>(g1_w,  g1t);
  wtrans_kernel<<<tg, 256, 0, stream>>>(g2_w,  g2t);
  wtrans_kernel<<<tg, 256, 0, stream>>>(fc3_w, fc3t);

  const int nelem = BN * D;
  f2bf_kernel<<<nelem / 256, 256, 0, stream>>>(sfeat, srcb, nelem);
  f2bf_kernel<<<nelem / 256, 256, 0, stream>>>(tfeat, tmpb, nelem);

  knn_kernel<<<BN, 256, 0, stream>>>(xyz, knn);

  const dim3 gg(BN / 16, D / 16);
  // sf = fc1(search), tf = fc1(template)  -> bf16
  gemm16_kernel<<<gg, 32, 0, stream>>>(srcb, fc1t, fc1_b, nullptr, nullptr, sfb, 0);
  gemm16_kernel<<<gg, 32, 0, stream>>>(tmpb, fc1t, fc1_b, nullptr, nullptr, tfb, 0);
  // q = tf@wq, k = sf@wk, v = sf@wv  -> fp32 (no bias)
  gemm16_kernel<<<gg, 32, 0, stream>>>(tfb, wqt, nullptr, nullptr, qmat, nullptr, 0);
  gemm16_kernel<<<gg, 32, 0, stream>>>(sfb, wkt, nullptr, nullptr, kall, nullptr, 0);
  gemm16_kernel<<<gg, 32, 0, stream>>>(sfb, wvt, nullptr, nullptr, vall, nullptr, 0);

  fused_attn_kernel<<<BN, 256, 0, stream>>>(xyz, knn, qmat, kall, vall,
                                            d1_w, d1_b, d2t, d2_b,
                                            g1t, g1_b, g2t, g2_b,
                                            attn_out, act);

  // res = act @ fc3 + fc3_b + search_feat
  gemm16_kernel<<<gg, 32, 0, stream>>>(act, fc3t, fc3_b, sfeat, res_out, nullptr, 0);
}
